// VectorQuantizer_69123203662177
// MI455X (gfx1250) — matrix operand order repaired and checked by reading, not compiled, not run on hardware
//
#include <hip/hip_runtime.h>

typedef __attribute__((ext_vector_type(16))) _Float16 v16h;
typedef __attribute__((ext_vector_type(8)))  float    v8f;
typedef __attribute__((ext_vector_type(4)))  unsigned int u32x4;
typedef __attribute__((ext_vector_type(8)))  int      i32x8;
typedef __attribute__((ext_vector_type(4)))  int      i32x4;

#define K_CODES 1024
#define DIM 64
#define N_ROWS (32 * 4096)          // 131072
#define NBLK 256
#define ROWS_PER_BLK (N_ROWS / NBLK)     // 512
#define MAIN_ITERS (ROWS_PER_BLK / 128)  // 4 (8 waves * 16 rows per iter)

// workspace byte offsets
#define WS_HI   0u
#define WS_LO   131072u             // 64 tiles * 2 frags * 32 lanes * 32B
#define WS_ESQ  262144u             // 1024 floats, TRANSPOSED: esqT[m][t] = e_sq[t*16+m]
#define WS_PART 266240u             // NBLK floats
#define STAGE_BYTES 266240u         // hi + lo + esqT copied to LDS
// LDS layout mirrors ws [0, STAGE_BYTES) plus 8 wave-sum floats
#define LDS_WSUM 266240u
#define LDS_TOTAL 266272u

#define WMMA_F16(A, B, C) \
  __builtin_amdgcn_wmma_f32_16x16x32_f16(false, (A), false, (B), (short)0, (C), false, false)

// ---------------------------------------------------------------------------
// Prep: build f16 hi/lo codebook tables in WMMA-B fragment layout + e_sq^T.
// B (32x16 f16) layout used here (contiguous k, same as A): lane l holds column N=l%16,
// K = (l/16)*16 + j for j=0..15 (16 contiguous f16 = 32B per lane).
// ---------------------------------------------------------------------------
__global__ void vq_prep(const float* __restrict__ E, unsigned char* __restrict__ ws) {
  const int b = blockIdx.x, tid = threadIdx.x;
  if (b < 16) {
    const int idx  = b * 256 + tid;        // tuple in [0, 4096) = t*64 + frag*32 + lane
    const int t    = idx >> 6;
    const int frag = (idx >> 5) & 1;
    const int lane = idx & 31;
    const int code = (t << 4) + (lane & 15);
    const int kb   = frag * 32 + ((lane >> 4) << 4);
    const float* src = E + code * DIM + kb;
    v16h h, l;
#pragma unroll
    for (int j = 0; j < 16; ++j) {
      float v = src[j];
      _Float16 hv = (_Float16)v;
      h[j] = hv;
      l[j] = (_Float16)(v - (float)hv);
    }
    ((v16h*)(ws + WS_HI))[idx] = h;
    ((v16h*)(ws + WS_LO))[idx] = l;
  } else {
    // e_sq transposed: esqT[m*64 + t] = sum_k E[t*16+m][k]^2
    float* esqT = (float*)(ws + WS_ESQ);
    for (int c = tid * 4; c < tid * 4 + 4; ++c) {
      const float* src = E + c * DIM;
      float s = 0.f;
#pragma unroll
      for (int j = 0; j < DIM; ++j) s += src[j] * src[j];
      const int t = c >> 4, m = c & 15;
      esqT[m * 64 + t] = s;
    }
  }
}

// ---------------------------------------------------------------------------
// Main: stage codebook to LDS via TDM, then WMMA distance + argmin + gather.
// ---------------------------------------------------------------------------
__global__ __launch_bounds__(256) void vq_main(const float* __restrict__ X,
                                               const float* __restrict__ E,
                                               const unsigned char* __restrict__ ws,
                                               float* __restrict__ out,
                                               float* __restrict__ partials) {
  extern __shared__ char smem[];
  const int tid  = threadIdx.x;
  const int wid  = tid >> 5;
  const int lane = tid & 31;
  const int m    = lane & 15;
  const int half = lane >> 4;

  // ---- Stage hi/lo/esqT tables (260KB) into LDS with one TDM op (wave 0) ----
  if (wid == 0) {
    unsigned long long ga = (unsigned long long)(const void*)ws;
    const unsigned int n8 = STAGE_BYTES / 8;  // 33280 8-byte units
    u32x4 g0;
    g0[0] = 1u;                                            // count=1 (valid D#)
    g0[1] = 0u;                                            // lds_addr = 0
    g0[2] = (unsigned int)(ga & 0xFFFFFFFFu);              // global_addr[31:0]
    g0[3] = (unsigned int)((ga >> 32) & 0x1FFFFFFu) | (2u << 30);  // addr[56:32] | type=2
    i32x8 g1;
    g1[0] = (int)(3u << 16);                               // data_size = 8B
    g1[1] = (int)((n8 & 0xFFFFu) << 16);                   // tensor_dim0 lo16
    g1[2] = (int)(((n8 >> 16) & 0xFFFFu) | (1u << 16));    // tensor_dim0 hi16, tensor_dim1=1
    g1[3] = (int)((n8 & 0xFFFFu) << 16);                   // tile_dim0
    g1[4] = 0;                                             // tile_dim1/2 unused
    g1[5] = (int)n8;                                       // tensor_dim0_stride lo32
    g1[6] = 0;
    g1[7] = 0;
    i32x4 gz = {0, 0, 0, 0};
#if defined(__clang_major__) && (__clang_major__ >= 23)
    i32x8 gz8 = {0, 0, 0, 0, 0, 0, 0, 0};
    __builtin_amdgcn_tensor_load_to_lds(g0, g1, gz, gz, gz8, 0);
#else
    __builtin_amdgcn_tensor_load_to_lds(g0, g1, gz, gz, 0);
#endif
    __builtin_amdgcn_s_wait_tensorcnt((short)0);
  }
  __syncthreads();

  const v16h*  Bh    = (const v16h*)(smem + WS_HI);
  const v16h*  Bl    = (const v16h*)(smem + WS_LO);
  const float* sEsqT = (const float*)(smem + WS_ESQ);
  float* sWsum = (float*)(smem + LDS_WSUM);

  float acc = 0.f;

  for (int it = 0; it < MAIN_ITERS; ++it) {
    const int row0 = blockIdx.x * ROWS_PER_BLK + it * 128 + wid * 16;
    const float* xrow = X + (size_t)(row0 + m) * DIM;
    if (it + 1 < MAIN_ITERS) __builtin_prefetch(xrow + 128 * DIM, 0, 1);

    // ---- Build A fragments (16x32 f16), hi + residual lo, and row |x|^2 ----
    // A layout: lane l holds M=l%16; frag f: K = f*32 + (l/16)*16 + j.
    v16h Ah[2], Al[2];
    float ssq = 0.f;
#pragma unroll
    for (int f = 0; f < 2; ++f) {
      const int k0 = f * 32 + half * 16;
      float4 p0 = *(const float4*)(xrow + k0);
      float4 p1 = *(const float4*)(xrow + k0 + 4);
      float4 q0 = *(const float4*)(xrow + k0 + 8);
      float4 q1 = *(const float4*)(xrow + k0 + 12);
      float vals[16] = {p0.x, p0.y, p0.z, p0.w, p1.x, p1.y, p1.z, p1.w,
                        q0.x, q0.y, q0.z, q0.w, q1.x, q1.y, q1.z, q1.w};
#pragma unroll
      for (int j = 0; j < 16; ++j) {
        float v = vals[j];
        ssq += v * v;
        _Float16 hv = (_Float16)v;
        Ah[f][j] = hv;
        Al[f][j] = (_Float16)(v - (float)hv);
      }
    }
    float xsq = ssq + __shfl_xor(ssq, 16, 32);   // combine the two K-halves of a row
    float xs8[8];
#pragma unroll
    for (int v = 0; v < 8; ++v) xs8[v] = __shfl(xsq, v + half * 24, 32);

    float mind[8];
    int   midx[8];
#pragma unroll
    for (int v = 0; v < 8; ++v) { mind[v] = 3.4e38f; midx[v] = 0; }

    // ---- 64 code tiles, 2 per pass: dot = Ah*Bh + Ah*Bl + Al*Bh, four
    //      independent WMMA accumulator chains to avoid RAW hazard NOPs ----
    for (int t = 0; t < 64; t += 2) {
      float2 e2 = *(const float2*)(sEsqT + m * 64 + t);   // e_sq for tiles t, t+1
      const int base = t * 64;  // v16h index of tile t frag0
      v16h bh0 = Bh[base + lane];
      v16h bh1 = Bh[base + 32 + lane];
      v16h bl0 = Bl[base + lane];
      v16h bl1 = Bl[base + 32 + lane];
      v16h ch0 = Bh[base + 64 + lane];
      v16h ch1 = Bh[base + 96 + lane];
      v16h cl0 = Bl[base + 64 + lane];
      v16h cl1 = Bl[base + 96 + lane];

      v8f a0 = {0.f,0.f,0.f,0.f,0.f,0.f,0.f,0.f};
      v8f a1 = {0.f,0.f,0.f,0.f,0.f,0.f,0.f,0.f};
      v8f b0 = {0.f,0.f,0.f,0.f,0.f,0.f,0.f,0.f};
      v8f b1 = {0.f,0.f,0.f,0.f,0.f,0.f,0.f,0.f};
      // round-robin across the 4 chains: no back-to-back dependent WMMAs
      a0 = WMMA_F16(Ah[0], bh0, a0);
      a1 = WMMA_F16(Ah[1], bh1, a1);
      b0 = WMMA_F16(Ah[0], ch0, b0);
      b1 = WMMA_F16(Ah[1], ch1, b1);
      a0 = WMMA_F16(Ah[0], bl0, a0);
      a1 = WMMA_F16(Ah[1], bl1, a1);
      b0 = WMMA_F16(Ah[0], cl0, b0);
      b1 = WMMA_F16(Ah[1], cl1, b1);
      a0 = WMMA_F16(Al[0], bh0, a0);
      a1 = WMMA_F16(Al[1], bh1, a1);
      b0 = WMMA_F16(Al[0], ch0, b0);
      b1 = WMMA_F16(Al[1], ch1, b1);

      const int code0 = (t << 4) + m;
      const int code1 = code0 + 16;
#pragma unroll
      for (int v = 0; v < 8; ++v) {
        float d0 = (xs8[v] - 2.0f * (a0[v] + a1[v])) + e2.x;  // reference association
        if (d0 < mind[v]) { mind[v] = d0; midx[v] = code0; }
      }
#pragma unroll
      for (int v = 0; v < 8; ++v) {
        float d1 = (xs8[v] - 2.0f * (b0[v] + b1[v])) + e2.y;
        if (d1 < mind[v]) { mind[v] = d1; midx[v] = code1; }
      }
    }

    // ---- argmin all-reduce across the 16 lanes of each half (first-min ties) ----
#pragma unroll
    for (int off = 1; off < 16; off <<= 1) {
#pragma unroll
      for (int v = 0; v < 8; ++v) {
        float od = __shfl_xor(mind[v], off, 32);
        int   oi = __shfl_xor(midx[v], off, 32);
        if (od < mind[v] || (od == mind[v] && oi < midx[v])) { mind[v] = od; midx[v] = oi; }
      }
    }

    // loss: min distance IS ||x - e_best||^2 (distance already includes x_sq)
    if (m == 0) {
#pragma unroll
      for (int v = 0; v < 8; ++v) acc += mind[v];
    }

    // ---- gather + straight-through output: out = x + (e[best] - x) ----
    // lane L handles row (L>>1); its index is held locally as midx[(L>>1)&7]
    const int rsel = (lane >> 1) & 7;
    int rIdx = midx[0];
#pragma unroll
    for (int v = 1; v < 8; ++v) rIdx = (rsel == v) ? midx[v] : rIdx;

    const int rL   = row0 + (lane >> 1);
    const int col0 = (lane & 1) * 32;
    const float* ep = E + (size_t)rIdx * DIM + col0;
    const float* xp = X + (size_t)rL * DIM + col0;
    float* op = out + (size_t)rL * DIM + col0;
#pragma unroll
    for (int i = 0; i < 8; ++i) {
      float4 ev = ((const float4*)ep)[i];
      float4 xv = ((const float4*)xp)[i];
      float4 ov;
      ov.x = xv.x + (ev.x - xv.x);
      ov.y = xv.y + (ev.y - xv.y);
      ov.z = xv.z + (ev.z - xv.z);
      ov.w = xv.w + (ev.w - xv.w);
      ((float4*)op)[i] = ov;
    }
  }

  // ---- per-block deterministic loss partial ----
  float wsum = acc + __shfl_xor(acc, 16, 32);
  if (lane == 0) sWsum[wid] = wsum;
  __syncthreads();
  if (tid == 0) {
    float s = 0.f;
#pragma unroll
    for (int w = 0; w < 8; ++w) s += sWsum[w];
    partials[blockIdx.x] = s;
  }
}

// ---------------------------------------------------------------------------
// Finalize: deterministic tree reduction of partials -> loss scalar.
// ---------------------------------------------------------------------------
__global__ void vq_finalize(const float* __restrict__ partials, float* __restrict__ out_loss) {
  __shared__ float s[256];
  const int tid = threadIdx.x;
  s[tid] = partials[tid];
  __syncthreads();
  for (int stride = 128; stride > 0; stride >>= 1) {
    if (tid < stride) s[tid] += s[tid + stride];
    __syncthreads();
  }
  if (tid == 0) out_loss[0] = s[0] * (1.25f / 8388608.0f);  // 1.25 * S / (N*D)
}

extern "C" void kernel_launch(void* const* d_in, const int* in_sizes, int n_in,
                              void* d_out, int out_size, void* d_ws, size_t ws_size,
                              hipStream_t stream) {
  (void)in_sizes; (void)n_in; (void)out_size; (void)ws_size;
  const float* x = (const float*)d_in[0];
  const float* E = (const float*)d_in[1];
  float* out = (float*)d_out;
  unsigned char* ws = (unsigned char*)d_ws;

  // allow >64KB dynamic LDS for the main kernel (no-op if already set)
  (void)hipFuncSetAttribute((const void*)vq_main,
                            hipFuncAttributeMaxDynamicSharedMemorySize, LDS_TOTAL);

  vq_prep<<<17, 256, 0, stream>>>(E, ws);
  vq_main<<<NBLK, 256, LDS_TOTAL, stream>>>(x, E, ws, out, (float*)(ws + WS_PART));
  vq_finalize<<<1, 256, 0, stream>>>((const float*)(ws + WS_PART), out + 8388608ull);
}
